// Transformer_21784074125362
// MI455X (gfx1250) — compile-verified, matrix operand order checked by reading, not run on hardware
//
#include <hip/hip_runtime.h>
#include <hip/hip_bf16.h>

typedef __attribute__((ext_vector_type(16))) __bf16 v16bf;
typedef __attribute__((ext_vector_type(8)))  __bf16 v8bf;
typedef __attribute__((ext_vector_type(4)))  __bf16 v4bf;
typedef __attribute__((ext_vector_type(8)))  float  v8f;
typedef __attribute__((ext_vector_type(4)))  float  f32x4;
typedef __attribute__((ext_vector_type(4)))  unsigned int u32x4;
typedef __attribute__((ext_vector_type(8)))  int    i32x8;
typedef __attribute__((ext_vector_type(4)))  int    i32x4;

#define D_MODEL 512
#define SEQ     512
#define BATCH   4
#define HEADS   8
#define DEPTH   64
#define DFF     2048
#define ROWS    (BATCH*SEQ)   /* 2048 */
#define VOCAB   32000
#define NEGBIG  (-1e9f)

#define LDS_PITCH 40          /* bf16 elems: 80B rows, bank-clean, 16B aligned */
#define PANEL     5120        /* 128*40 bf16 elems = 10240 B per panel */

#if defined(__has_builtin)
#if __has_builtin(__builtin_amdgcn_tensor_load_to_lds) && __has_builtin(__builtin_amdgcn_s_wait_tensorcnt)
#define HAVE_TDM 1
#endif
#endif

union FragU { v16bf v; v8bf h[2]; };

// ---------------------------------------------------------------------------
// Batched GEMM: C = act(alpha * A x B + bias)
// A fp32 [M,K] (lda).  B: either pre-transposed bf16 Bt[N][K] (TDM-fed) or
// fp32 with element (k,n) at k*ldbk + n*ldbn.  bf16 WMMA 16x16x32, fp32 acc.
// Block tile 128x128, BK=32, double-buffered LDS (A0|A1|B0|B1 = 40KB),
// software pipeline: next-tile global loads + TDM DMA overlap current WMMAs.
// 8 wave32/block; each wave owns 32x64 = 2x4 fragments. M % 128 == 0.
// ---------------------------------------------------------------------------
__global__ __launch_bounds__(256) void gemm_wmma_bf16(
    const float* __restrict__ A, const float* __restrict__ B,
    const __bf16* __restrict__ Bt,
    const float* __restrict__ bias, float* __restrict__ C,
    int M, int N, int K, int lda, long long ldbk, long long ldbn, int ldc,
    long long sAb, long long sAh, long long sBb, long long sBh,
    long long sCb, long long sCh, int nH, float alpha, int relu)
{
  extern __shared__ __align__(16) char smem[];
  __bf16* base = (__bf16*)smem;
  // LDS byte offsets: A0=0, A1=10240, B0=20480, B1=30720

  const int tid  = threadIdx.x;
  const int lane = tid & 31;
  const int wid  = tid >> 5;
  const int wm   = wid & 3;
  const int wn   = wid >> 2;
  const int half = lane >> 4;
  const int lr   = lane & 15;

  const int bm = blockIdx.y * 128;
  const int bn = blockIdx.x * 128;
  const int z  = blockIdx.z;
  const int bb = z / nH;
  const int hh = z - bb * nH;

  const float* Ab = A + (long long)bb * sAb + (long long)hh * sAh;
  const float* Bb = B + (long long)bb * sBb + (long long)hh * sBh;
  float*       Cb = C + (long long)bb * sCb + (long long)hh * sCh;

  v8f zero = {0.f,0.f,0.f,0.f,0.f,0.f,0.f,0.f};
  v8f acc[2][4];
  #pragma unroll
  for (int i = 0; i < 2; ++i)
    #pragma unroll
    for (int j = 0; j < 4; ++j) acc[i][j] = zero;

  f32x4 va[4];   // staged A chunks (16 fp32/thread)
  f32x4 vb[4];   // staged fp32 B chunks
  v8bf  vbt[2];  // staged bf16 B chunks (no-TDM fallback)

  // ---- load phase: issue all global loads for tile (kk), regs only --------
  auto loadA = [&](int kk) {
    #pragma unroll
    for (int j = 0; j < 4; ++j) {
      int ci = tid + (j << 8);
      int r = ci >> 3, c4 = (ci & 7) << 2;
      va[j] = *(const f32x4*)(Ab + (long long)(bm + r) * lda + (kk + c4));
    }
  };
  auto loadB = [&](int kk) {
    if (ldbk == 1) {            // K-contiguous (Q.K^T); N multiple of 128 here
      #pragma unroll
      for (int j = 0; j < 4; ++j) {
        int ci = tid + (j << 8);
        int n = ci >> 3, c4 = (ci & 7) << 2;
        vb[j] = *(const f32x4*)(Bb + (long long)(bn + n) * ldbn + (kk + c4));
      }
    } else {                    // N-contiguous (P.V); N may be 64
      #pragma unroll
      for (int j = 0; j < 4; ++j) {
        int ci = tid + (j << 8);
        int k = ci >> 5, n4 = (ci & 31) << 2;
        int gn = bn + n4;
        f32x4 zz = {0.f,0.f,0.f,0.f};
        vb[j] = (gn < N) ? *(const f32x4*)(Bb + (long long)(kk + k) * ldbk + gn) : zz;
      }
    }
  };
#ifndef HAVE_TDM
  auto loadBt = [&](int kk) {
    #pragma unroll
    for (int j = 0; j < 2; ++j) {
      int ci = tid + (j << 8);
      int n = ci >> 2, c = ci & 3;
      vbt[j] = *(const v8bf*)(Bt + (long long)(bn + n) * K + kk + c*8);
    }
  };
#endif
  // TDM: DMA 128(n) x 32(k) bf16 panel of Bt[N][K] into LDS buffer `buf`,
  // hardware-padding 16B after every 64B row -> 80B pitch.
  auto tdmB = [&](int buf, int kk) {
#ifdef HAVE_TDM
    if (wid == 0) {
      unsigned long long ga =
          (unsigned long long)(const void*)(Bt + (long long)bn * K + kk);
      u32x4 g0;
      g0.x = 1u;                                     // count = 1
      g0.y = 20480u + (unsigned)buf * 10240u;        // lds_addr
      g0.z = (unsigned)ga;
      g0.w = (unsigned)((ga >> 32) & 0x1FFFFFFu) | (2u << 30);   // hi | type=2
      unsigned td0 = (unsigned)(K - kk);
      unsigned td1 = (unsigned)(N - bn);
      i32x8 g1;
      g1[0] = (int)((1u << 16) | (1u << 20) | (3u << 22) | (3u << 25));
               // data_size=2B | pad_enable | pad_interval=16DW | pad_amount=4DW
      g1[1] = (int)((td0 & 0xFFFFu) << 16);
      g1[2] = (int)((td0 >> 16) | ((td1 & 0xFFFFu) << 16));
      g1[3] = (int)((td1 >> 16) | (32u << 16));      // tile_dim0 = 32
      g1[4] = (int)128;                              // tile_dim1 = 128
      g1[5] = K;                                     // tensor_dim0_stride
      g1[6] = 0; g1[7] = 0;
      i32x4 z4 = {0,0,0,0};
#if __clang_major__ >= 23
      i32x8 z8 = {0,0,0,0,0,0,0,0};
      __builtin_amdgcn_tensor_load_to_lds(g0, g1, z4, z4, z8, 0);
#else
      __builtin_amdgcn_tensor_load_to_lds(g0, g1, z4, z4, 0);
#endif
    }
#endif
  };
  auto tdmWait = [&]() {
#ifdef HAVE_TDM
    if (wid == 0) __builtin_amdgcn_s_wait_tensorcnt((short)0);
#endif
  };

  // ---- store phase: convert regs -> bf16 LDS panels -----------------------
  auto storeA = [&](int buf) {
    __bf16* As = base + buf * PANEL;
    #pragma unroll
    for (int j = 0; j < 4; ++j) {
      int ci = tid + (j << 8);
      int r = ci >> 3, c4 = (ci & 7) << 2;
      v4bf o; o.x=(__bf16)va[j].x; o.y=(__bf16)va[j].y; o.z=(__bf16)va[j].z; o.w=(__bf16)va[j].w;
      *(v4bf*)(As + r*LDS_PITCH + c4) = o;
    }
  };
  auto storeB = [&](int buf) {
    __bf16* Bs = base + (2 + buf) * PANEL;
    if (ldbk == 1) {
      #pragma unroll
      for (int j = 0; j < 4; ++j) {
        int ci = tid + (j << 8);
        int n = ci >> 3, c4 = (ci & 7) << 2;
        v4bf o; o.x=(__bf16)vb[j].x; o.y=(__bf16)vb[j].y; o.z=(__bf16)vb[j].z; o.w=(__bf16)vb[j].w;
        *(v4bf*)(Bs + n*LDS_PITCH + c4) = o;
      }
    } else {
      #pragma unroll
      for (int j = 0; j < 4; ++j) {
        int ci = tid + (j << 8);
        int k = ci >> 5, n4 = (ci & 31) << 2;
        Bs[(n4+0)*LDS_PITCH + k] = (__bf16)vb[j].x;
        Bs[(n4+1)*LDS_PITCH + k] = (__bf16)vb[j].y;
        Bs[(n4+2)*LDS_PITCH + k] = (__bf16)vb[j].z;
        Bs[(n4+3)*LDS_PITCH + k] = (__bf16)vb[j].w;
      }
    }
  };
#ifndef HAVE_TDM
  auto storeBt = [&](int buf) {
    __bf16* Bs = base + (2 + buf) * PANEL;
    #pragma unroll
    for (int j = 0; j < 2; ++j) {
      int ci = tid + (j << 8);
      int n = ci >> 2, c = ci & 3;
      *(v8bf*)(Bs + n*LDS_PITCH + c*8) = vbt[j];
    }
  };
#endif

  auto stageLoad = [&](int buf, int kk) {
    loadA(kk);
    if (Bt) {
#ifdef HAVE_TDM
      tdmB(buf, kk);
#else
      loadBt(kk);
#endif
    } else {
      loadB(kk);
    }
  };
  auto stageStore = [&](int buf) {
    storeA(buf);
    if (Bt) {
#ifndef HAVE_TDM
      storeBt(buf);
#endif
      tdmWait();
    } else {
      storeB(buf);
    }
  };

  // ---- compute: fragments (ds_load_b128 pairs, ISA layout) + 8 WMMAs ------
  auto compute = [&](int buf) {
    const __bf16* As = base + buf * PANEL;
    const __bf16* Bs = base + (2 + buf) * PANEL;
    FragU af[2], bf[4];
    #pragma unroll
    for (int mi = 0; mi < 2; ++mi) {
      const __bf16* p = &As[(wm*32 + mi*16 + lr) * LDS_PITCH];
      af[mi].h[0] = *(const v8bf*)(p + half*8);
      af[mi].h[1] = *(const v8bf*)(p + 16 + half*8);
    }
    #pragma unroll
    for (int ni = 0; ni < 4; ++ni) {
      const __bf16* p = &Bs[(wn*64 + ni*16 + lr) * LDS_PITCH];
      bf[ni].h[0] = *(const v8bf*)(p + half*8);
      bf[ni].h[1] = *(const v8bf*)(p + 16 + half*8);
    }
    #pragma unroll
    for (int mi = 0; mi < 2; ++mi)
      #pragma unroll
      for (int ni = 0; ni < 4; ++ni)
        acc[mi][ni] = __builtin_amdgcn_wmma_f32_16x16x32_bf16(
            false, af[mi].v, false, bf[ni].v, (short)0, acc[mi][ni], false, false);
  };

  // ---- pipelined main loop ------------------------------------------------
  stageLoad(0, 0);
  stageStore(0);
  __syncthreads();

  int pp = 0;
  for (int k0 = 0; k0 < K; k0 += 32) {
    bool more = (k0 + 32) < K;
    if (more) stageLoad(pp ^ 1, k0 + 32);          // global loads + TDM issue
    if (k0 + 64 < K) {                              // prefetch 2 tiles ahead
      int r  = tid >> 1;
      int cc = (tid & 1) << 4;
      __builtin_prefetch(&Ab[(long long)(bm + r) * lda + (k0 + 64 + cc)], 0, 1);
    }
    compute(pp);                                    // WMMAs overlap DMA/loads
    if (more) {
      stageStore(pp ^ 1);                           // cvt+LDS store, TDM wait
      __syncthreads();
    }
    pp ^= 1;
  }

  // ---- epilogue: lanes 0-15: N=lr, M=r; lanes 16-31: M=8+r ----------------
  #pragma unroll
  for (int ni = 0; ni < 4; ++ni) {
    int col = bn + wn*64 + ni*16 + lr;
    if (col >= N) continue;
    float bv = bias ? bias[col] : 0.f;
    #pragma unroll
    for (int mi = 0; mi < 2; ++mi) {
      #pragma unroll
      for (int r = 0; r < 8; ++r) {
        int m = bm + wm*32 + mi*16 + half*8 + r;
        float cv = acc[mi][ni][r] * alpha + bv;
        if (relu) cv = fmaxf(cv, 0.f);
        Cb[(long long)m * ldc + col] = cv;
      }
    }
  }
}

// ---------------------------------------------------------------------------
// Weight convert+transpose: W fp32 [K,N] -> Wt bf16 [N,K]  (LDS 32x32 tiles)
// ---------------------------------------------------------------------------
__global__ __launch_bounds__(256) void wconv_kernel(
    const float* __restrict__ Wsrc, __bf16* __restrict__ Wt, int Kd, int N)
{
  __shared__ __bf16 t[32*33];
  int n0 = blockIdx.x * 32, k0 = blockIdx.y * 32;
  int tx = threadIdx.x & 31, ty = threadIdx.x >> 5;
  #pragma unroll
  for (int j = 0; j < 4; ++j) {
    int kl = ty + j*8;
    t[kl*33 + tx] = (__bf16)Wsrc[(long long)(k0 + kl) * N + n0 + tx];
  }
  __syncthreads();
  #pragma unroll
  for (int j = 0; j < 4; ++j) {
    int nl = ty + j*8;
    Wt[(long long)(n0 + nl) * Kd + k0 + tx] = t[tx*33 + nl];
  }
}

// ---------------------------------------------------------------------------
// Embedding gather * sqrt(d_model) + positional encoding
// ---------------------------------------------------------------------------
__global__ __launch_bounds__(256) void embed_kernel(
    const int* __restrict__ tok, const float* __restrict__ emb,
    const float* __restrict__ pe, float* __restrict__ x)
{
  int row = blockIdx.x;
  int s = row & (SEQ - 1);
  int t = tok[row];
  const float scale = 22.627416997969522f;
  #pragma unroll
  for (int j = 0; j < 2; ++j) {
    int c = threadIdx.x + j*256;
    x[(long long)row*D_MODEL + c] = emb[(long long)t*D_MODEL + c]*scale + pe[s*D_MODEL + c];
  }
}

// ---------------------------------------------------------------------------
// Masked softmax over rows of [B*H*SEQ, SEQ]; masks synthesized from tokens.
// ---------------------------------------------------------------------------
__global__ __launch_bounds__(256) void softmax_kernel(
    float* __restrict__ S, const int* __restrict__ tok, int causal)
{
  __shared__ float red[256];
  int row = blockIdx.x;
  int q   = row & (SEQ - 1);
  int b   = (row >> 9) >> 3;
  float* Sr = S + (long long)row * SEQ;
  const int* tk = tok + b * SEQ;

  float v[2]; int cs[2];
  float mx = -3.4e38f;
  #pragma unroll
  for (int j = 0; j < 2; ++j) {
    int c = threadIdx.x + j*256;
    cs[j] = c;
    float x = Sr[c];
    bool masked = (tk[c] == 0) || (causal && (c > q));
    x += masked ? NEGBIG : 0.f;
    v[j] = x;
    mx = fmaxf(mx, x);
  }
  red[threadIdx.x] = mx; __syncthreads();
  for (int s2 = 128; s2 > 0; s2 >>= 1) {
    if (threadIdx.x < s2) red[threadIdx.x] = fmaxf(red[threadIdx.x], red[threadIdx.x + s2]);
    __syncthreads();
  }
  mx = red[0]; __syncthreads();
  float sum = 0.f;
  #pragma unroll
  for (int j = 0; j < 2; ++j) { v[j] = __expf(v[j] - mx); sum += v[j]; }
  red[threadIdx.x] = sum; __syncthreads();
  for (int s2 = 128; s2 > 0; s2 >>= 1) {
    if (threadIdx.x < s2) red[threadIdx.x] += red[threadIdx.x + s2];
    __syncthreads();
  }
  float inv = 1.f / red[0];
  #pragma unroll
  for (int j = 0; j < 2; ++j) Sr[cs[j]] = v[j] * inv;
}

// ---------------------------------------------------------------------------
// dst = LayerNorm(a + b) * g + be   (in-place safe)
// ---------------------------------------------------------------------------
__global__ __launch_bounds__(256) void add_ln_kernel(
    const float* __restrict__ a, const float* __restrict__ b,
    const float* __restrict__ g, const float* __restrict__ be,
    float* __restrict__ dst)
{
  __shared__ float r1[256], r2[256];
  int row = blockIdx.x;
  const float* ar = a + (long long)row * D_MODEL;
  const float* br = b + (long long)row * D_MODEL;
  int c0 = threadIdx.x, c1 = threadIdx.x + 256;
  float s0 = ar[c0] + br[c0];
  float s1 = ar[c1] + br[c1];
  r1[threadIdx.x] = s0 + s1;
  r2[threadIdx.x] = s0*s0 + s1*s1;
  __syncthreads();
  for (int s2 = 128; s2 > 0; s2 >>= 1) {
    if (threadIdx.x < s2) { r1[threadIdx.x] += r1[threadIdx.x+s2]; r2[threadIdx.x] += r2[threadIdx.x+s2]; }
    __syncthreads();
  }
  float mean = r1[0] * (1.f/512.f);
  float var  = r2[0] * (1.f/512.f) - mean*mean;
  float rstd = rsqrtf(var + 1e-6f);
  float* dr = dst + (long long)row * D_MODEL;
  dr[c0] = (s0 - mean) * rstd * g[c0] + be[c0];
  dr[c1] = (s1 - mean) * rstd * g[c1] + be[c1];
}

// ---------------------------------------------------------------------------
extern "C" void kernel_launch(void* const* d_in, const int* in_sizes, int n_in,
                              void* d_out, int out_size, void* d_ws, size_t ws_size,
                              hipStream_t stream)
{
  (void)in_sizes; (void)out_size; (void)ws_size;
  if (n_in < 176) return;

  int p = 0;
  const int* inp = (const int*)d_in[p++];
  const int* tar = (const int*)d_in[p++];
  auto F = [&]() { return (const float*)d_in[p++]; };

  struct DenseP { const float* b; const float* w; };
  struct LnP    { const float* b; const float* g; };
  struct MhaP   { DenseP wk, wo, wq, wv; };
  struct DecL   { DenseP w1, w2; LnP ln1, ln2, ln3; MhaP m1, m2; };
  struct EncL   { DenseP w1, w2; LnP ln1, ln2; MhaP m; };

  const float* dec_emb = F();
  DecL dl[4];
  for (int l = 0; l < 4; ++l) {
    dl[l].w1.b = F(); dl[l].w1.w = F();
    dl[l].w2.b = F(); dl[l].w2.w = F();
    dl[l].ln1.b = F(); dl[l].ln1.g = F();
    dl[l].ln2.b = F(); dl[l].ln2.g = F();
    dl[l].ln3.b = F(); dl[l].ln3.g = F();
    dl[l].m1.wk.b = F(); dl[l].m1.wk.w = F();
    dl[l].m1.wo.b = F(); dl[l].m1.wo.w = F();
    dl[l].m1.wq.b = F(); dl[l].m1.wq.w = F();
    dl[l].m1.wv.b = F(); dl[l].m1.wv.w = F();
    dl[l].m2.wk.b = F(); dl[l].m2.wk.w = F();
    dl[l].m2.wo.b = F(); dl[l].m2.wo.w = F();
    dl[l].m2.wq.b = F(); dl[l].m2.wq.w = F();
    dl[l].m2.wv.b = F(); dl[l].m2.wv.w = F();
  }
  const float* dec_pe  = F();
  const float* enc_emb = F();
  EncL el[4];
  for (int l = 0; l < 4; ++l) {
    el[l].w1.b = F(); el[l].w1.w = F();
    el[l].w2.b = F(); el[l].w2.w = F();
    el[l].ln1.b = F(); el[l].ln1.g = F();
    el[l].ln2.b = F(); el[l].ln2.g = F();
    el[l].m.wk.b = F(); el[l].m.wk.w = F();
    el[l].m.wo.b = F(); el[l].m.wo.w = F();
    el[l].m.wq.b = F(); el[l].m.wq.w = F();
    el[l].m.wv.b = F(); el[l].m.wv.w = F();
  }
  const float* enc_pe = F();
  const float* fin_b  = F();
  const float* fin_w  = F();

  const long long ACT = (long long)ROWS * D_MODEL;
  float* W  = (float*)d_ws;
  float* X  = W;
  float* R  = W +  1*ACT;
  float* Q  = W +  2*ACT;
  float* Kb = W +  3*ACT;
  float* V  = W +  4*ACT;
  float* T  = W +  5*ACT;
  float* E  = W +  6*ACT;
  float* H  = W +  7*ACT;            // ROWS x DFF (4*ACT)
  float* S  = W + 11*ACT;            // B*H x SEQ x SEQ (8*ACT)
  __bf16* WT = (__bf16*)(W + 19*ACT); // bf16 transposed weight panel (<=32MB)

  auto gemm = [&](const float* A, const float* B, const __bf16* Bt,
                  const float* bias, float* C,
                  int M, int N, int K, int lda, long long ldbk, long long ldbn, int ldc,
                  long long sAb, long long sAh, long long sBb, long long sBh,
                  long long sCb, long long sCh, int nB, int nH, float alpha, int relu) {
    dim3 g((N + 127)/128, (M + 127)/128, nB*nH);
    gemm_wmma_bf16<<<g, 256, 4*PANEL*2, stream>>>(
        A, B, Bt, bias, C, M, N, K, lda, ldbk, ldbn, ldc,
        sAb, sAh, sBb, sBh, sCb, sCh, nH, alpha, relu);
  };
  auto dense = [&](const float* A, DenseP dP, float* C, int N, int K, int relu) {
    wconv_kernel<<<dim3(N/32, K/32), 256, 0, stream>>>(dP.w, WT, K, N);
    gemm(A, dP.w, WT, dP.b, C, ROWS, N, K, K, N, 1, N,
         0,0,0,0,0,0, 1,1, 1.f, relu);
  };
  auto attn_scores = [&](const float* Qp, const float* Kp) {
    gemm(Qp, Kp, nullptr, nullptr, S, SEQ, SEQ, DEPTH, D_MODEL, 1, D_MODEL, SEQ,
         (long long)SEQ*D_MODEL, DEPTH, (long long)SEQ*D_MODEL, DEPTH,
         (long long)HEADS*SEQ*SEQ, (long long)SEQ*SEQ, BATCH, HEADS, 0.125f, 0);
  };
  auto attn_pv = [&](const float* Vp, float* Out) {
    gemm(S, Vp, nullptr, nullptr, Out, SEQ, DEPTH, SEQ, SEQ, D_MODEL, 1, D_MODEL,
         (long long)HEADS*SEQ*SEQ, (long long)SEQ*SEQ,
         (long long)SEQ*D_MODEL, DEPTH,
         (long long)SEQ*D_MODEL, DEPTH, BATCH, HEADS, 1.f, 0);
  };
  auto softmax = [&](const int* tok, int causal) {
    softmax_kernel<<<BATCH*HEADS*SEQ, 256, 0, stream>>>(S, tok, causal);
  };
  auto addln = [&](const float* a, const float* b2, LnP ln, float* dst) {
    add_ln_kernel<<<ROWS, 256, 0, stream>>>(a, b2, ln.g, ln.b, dst);
  };

  // ---------------- Encoder ----------------
  embed_kernel<<<ROWS, 256, 0, stream>>>(inp, enc_emb, enc_pe, X);
  for (int l = 0; l < 4; ++l) {
    dense(X, el[l].m.wq, Q,  D_MODEL, D_MODEL, 0);
    dense(X, el[l].m.wk, Kb, D_MODEL, D_MODEL, 0);
    dense(X, el[l].m.wv, V,  D_MODEL, D_MODEL, 0);
    attn_scores(Q, Kb);
    softmax(inp, 0);
    attn_pv(V, T);
    dense(T, el[l].m.wo, Q, D_MODEL, D_MODEL, 0);
    addln(Q, X, el[l].ln1, R);
    dense(R, el[l].w1, H, DFF, D_MODEL, 1);
    dense(H, el[l].w2, T, D_MODEL, DFF, 0);
    addln(T, R, el[l].ln2, (l == 3) ? E : X);
  }

  // ---------------- Decoder ----------------
  embed_kernel<<<ROWS, 256, 0, stream>>>(tar, dec_emb, dec_pe, X);
  for (int l = 0; l < 4; ++l) {
    dense(X, dl[l].m1.wq, Q,  D_MODEL, D_MODEL, 0);
    dense(X, dl[l].m1.wk, Kb, D_MODEL, D_MODEL, 0);
    dense(X, dl[l].m1.wv, V,  D_MODEL, D_MODEL, 0);
    attn_scores(Q, Kb);
    softmax(tar, 1);
    attn_pv(V, T);
    dense(T, dl[l].m1.wo, Q, D_MODEL, D_MODEL, 0);
    addln(Q, X, dl[l].ln1, R);

    dense(R, dl[l].m2.wq, Q,  D_MODEL, D_MODEL, 0);
    dense(E, dl[l].m2.wk, Kb, D_MODEL, D_MODEL, 0);
    dense(E, dl[l].m2.wv, V,  D_MODEL, D_MODEL, 0);
    attn_scores(Q, Kb);
    softmax(inp, 0);
    attn_pv(V, T);
    dense(T, dl[l].m2.wo, Q, D_MODEL, D_MODEL, 0);
    addln(Q, R, dl[l].ln2, X);

    dense(X, dl[l].w1, H, DFF, D_MODEL, 1);
    dense(H, dl[l].w2, T, D_MODEL, DFF, 0);
    addln(T, X, dl[l].ln3, X);
  }

  // ---------------- Final vocab projection ----------------
  wconv_kernel<<<dim3(VOCAB/32, D_MODEL/32), 256, 0, stream>>>(fin_w, WT, D_MODEL, VOCAB);
  gemm(X, fin_w, WT, fin_b, (float*)d_out, ROWS, VOCAB, D_MODEL, D_MODEL,
       (long long)VOCAB, 1, VOCAB, 0,0,0,0,0,0, 1,1, 1.f, 0);
}
